// HGTLayer_23897198035537
// MI455X (gfx1250) — matrix operand order repaired and checked by reading, not compiled, not run on hardware
//
#include <hip/hip_runtime.h>

// ---------------------------------------------------------------------------
// HGT layer for MI455X (gfx1250, wave32).
// bf16 WMMA GEMMs with async-to-LDS staging; fp32 softmax/scatter/LN.
// ---------------------------------------------------------------------------

typedef __attribute__((ext_vector_type(16))) __bf16 v16bf;
typedef __attribute__((ext_vector_type(8)))  float  v8f;

static constexpr int Dm  = 256;   // model dim
static constexpr int NH  = 8;     // heads
static constexpr int DKC = 32;    // head dim

// ---------------- CDNA5 async copy helpers ----------------

__device__ __forceinline__ void async_copy_b128(void* ldsPtr, const void* gPtr) {
  unsigned l = (unsigned)(uintptr_t)ldsPtr;   // low 32 bits of generic ptr = LDS offset
  asm volatile("global_load_async_to_lds_b128 %0, %1, off"
               :: "v"(l), "v"(gPtr) : "memory");
}
__device__ __forceinline__ void wait_async0() {
  asm volatile("s_wait_asynccnt 0x0" ::: "memory");
}

// ---------------- small utility kernels ----------------

__global__ void zero_f(float* p, long n) {
  long i = (long)blockIdx.x * blockDim.x + threadIdx.x;
  long stride = (long)gridDim.x * blockDim.x;
  for (; i < n; i += stride) p[i] = 0.f;
}

__global__ void init_softmax(float* smax, float* ssum, int n) {
  int i = blockIdx.x * blockDim.x + threadIdx.x;
  if (i < n) { smax[i] = __int_as_float(0xff800000); ssum[i] = 0.f; }
}

// fp32 activations -> bf16, columns kept in natural (contiguous) K order
// within each 32-block, matching the contiguous B-fragment order.
__global__ void cvt_perm_bf16(const float* __restrict__ in, __bf16* __restrict__ out, long n) {
  long i = (long)blockIdx.x * blockDim.x + threadIdx.x;
  long stride = (long)gridDim.x * blockDim.x;
  for (; i < n; i += stride) {
    int col = (int)(i & 31);
    int g = col >> 3;
    long slot = (long)(((g >> 1) << 4) | ((g & 1) << 3) | (col & 7));
    out[(i & ~31L) | slot] = (__bf16)in[i];
  }
}

// W[256,256] fp32 -> transposed bf16 Wt[col][k]
__global__ void cvt_w_t(const float* __restrict__ W, __bf16* __restrict__ Wt) {
  int idx = blockIdx.x * blockDim.x + threadIdx.x;   // 65536; idx = o*256 + i
  int o = idx >> 8, i = idx & 255;
  Wt[idx] = (__bf16)W[(size_t)i * Dm + o];
}

// Wt_eff[o, i] = sum_d W[i, h*32+d] * att[h, d, e]  with o = h*32+e (transposed bf16 out)
__global__ void compose_w_t(const float* __restrict__ W, const float* __restrict__ att,
                            __bf16* __restrict__ Wt) {
  int idx = blockIdx.x * blockDim.x + threadIdx.x;   // 65536; idx = o*256 + i
  int o = idx >> 8, i = idx & 255;
  int h = o >> 5, e = o & 31;
  const float* wrow = W + (size_t)i * Dm + h * DKC;
  const float* acol = att + h * (DKC * DKC) + e;     // stride 32 over d
  float s = 0.f;
#pragma unroll
  for (int d = 0; d < DKC; d++) s += wrow[d] * acol[d * DKC];
  Wt[idx] = (__bf16)s;
}

// b_eff[h*32+e] = sum_d b[h*32+d] * att[h, d, e]
__global__ void compose_b(const float* __restrict__ b, const float* __restrict__ att,
                          float* __restrict__ bout) {
  int o = threadIdx.x;               // 256 threads
  int h = o >> 5, e = o & 31;
  float s = 0.f;
#pragma unroll
  for (int d = 0; d < DKC; d++) s += b[h * DKC + d] * att[h * (DKC * DKC) + d * DKC + e];
  bout[o] = s;
}

// ---------------- WMMA GEMM:  C[M,256] = A_bf16[M,256] @ W + bias ----------------
// A is bf16 in natural K-order (rows padded to a multiple of 128).
// Wt is bf16 transposed [col][k]. Staging is a pure async byte-copy into LDS,
// double buffered so tile k+1 streams in while tile k is in the WMMAs.
// block = 256 threads (8 waves); block tile 128(M) x 64(N); K-step 32.

__global__ __launch_bounds__(256) void gemm_bf16(const __bf16* __restrict__ A,
                                                 const __bf16* __restrict__ Wt,
                                                 const float* __restrict__ bias,
                                                 float* __restrict__ C, int M) {
  __shared__ __attribute__((aligned(32))) __bf16 lA[2][128 * 32];
  __shared__ __attribute__((aligned(32))) __bf16 lB[2][64 * 32];

  const int tid   = threadIdx.x;
  const int lane  = tid & 31;
  const int wid   = tid >> 5;
  const int waveM = wid & 3;     // 0..3 -> 32-row slab
  const int waveN = wid >> 2;    // 0..1 -> 32-col slab
  const int rowBase = blockIdx.y * 128;
  const int c0      = blockIdx.x * 64;

  v8f acc[2][2] = {};

  auto issueTile = [&](int i) {
    const int k0  = i * 32;
    const int buf = i & 1;
    // A tile: 128 rows x 64B  = 512 b128 units (2 per thread)
#pragma unroll
    for (int it = 0; it < 2; it++) {
      int u = tid + it * 256;
      int r = u >> 2, q = u & 3;
      async_copy_b128(&lA[buf][r * 32 + q * 8],
                      A + (size_t)(rowBase + r) * Dm + k0 + q * 8);
    }
    // B tile: 64 cols x 64B = 256 b128 units (1 per thread)
    {
      int c = tid >> 2, q = tid & 3;
      async_copy_b128(&lB[buf][c * 32 + q * 8],
                      Wt + (size_t)(c0 + c) * Dm + k0 + q * 8);
    }
  };

  issueTile(0);
  for (int i = 0; i < 8; i++) {
    wait_async0();
    __syncthreads();           // tile i visible to all waves; compute(i-1) finished
    if (i < 7) issueTile(i + 1);

    const int buf  = i & 1;
    const int half = (lane >> 4) << 4;   // lanes 0-15: K-half 0, 16-31: K-half 1
    v16bf af[2], bfv[2];
#pragma unroll
    for (int wm = 0; wm < 2; wm++) {
      int arow = waveM * 32 + wm * 16 + (lane & 15);
      af[wm] = *(const v16bf*)&lA[buf][arow * 32 + half];
    }
#pragma unroll
    for (int wn = 0; wn < 2; wn++) {
      int bcol = waveN * 32 + wn * 16 + (lane & 15);
      bfv[wn] = *(const v16bf*)&lB[buf][bcol * 32 + half];
    }
#pragma unroll
    for (int wm = 0; wm < 2; wm++)
#pragma unroll
      for (int wn = 0; wn < 2; wn++)
        acc[wm][wn] = __builtin_amdgcn_wmma_f32_16x16x32_bf16(
            false, af[wm], false, bfv[wn], (short)0, acc[wm][wn], false, false);
  }

  // epilogue: C/D layout VGPR r -> M = r + 8*(lane>=16), N = lane&15
#pragma unroll
  for (int wm = 0; wm < 2; wm++) {
#pragma unroll
    for (int wn = 0; wn < 2; wn++) {
      int trow = rowBase + waveM * 32 + wm * 16 + ((lane >> 4) << 3);
      int tcol = c0 + waveN * 32 + wn * 16 + (lane & 15);
      float bv = bias[tcol];
#pragma unroll
      for (int r2 = 0; r2 < 8; r2++) {
        int gr = trow + r2;
        if (gr < M) C[(size_t)gr * Dm + tcol] = acc[wm][wn][r2] + bv;
      }
    }
  }
}

// ---------------- edge kernels ----------------

__device__ __forceinline__ void atomicMaxF(float* addr, float val) {
  if (val >= 0.f) atomicMax((int*)addr, __float_as_int(val));
  else            atomicMin((unsigned int*)addr, __float_as_uint(val));
}

__global__ void edge_score(const float* __restrict__ Q, const float* __restrict__ K,
                           const int* __restrict__ src, const int* __restrict__ dst,
                           const float* __restrict__ pri, float* __restrict__ score,
                           float* __restrict__ smax, int EH) {
  int t = blockIdx.x * blockDim.x + threadIdx.x;
  if (t >= EH) return;
  int e = t >> 3, h = t & 7;
  int s = src[e], d = dst[e];
  const float4* q4 = (const float4*)(Q + (size_t)d * Dm + h * DKC);
  const float4* k4 = (const float4*)(K + (size_t)s * Dm + h * DKC);
  float acc = 0.f;
#pragma unroll
  for (int i = 0; i < 8; i++) {
    float4 a = q4[i], b = k4[i];
    acc += a.x * b.x + a.y * b.y + a.z * b.z + a.w * b.w;
  }
  acc *= pri[h] * 0.17677669529663687f;   // pri / sqrt(32)
  score[t] = acc;
  atomicMaxF(&smax[(size_t)d * NH + h], acc);
}

__global__ void edge_exp(const float* __restrict__ score, const float* __restrict__ smax,
                         const int* __restrict__ dst, float* __restrict__ ex,
                         float* __restrict__ ssum, int EH) {
  int t = blockIdx.x * blockDim.x + threadIdx.x;
  if (t >= EH) return;
  int e = t >> 3, h = t & 7;
  int d = dst[e];
  float v = __expf(score[t] - smax[(size_t)d * NH + h]);
  ex[t] = v;
  atomicAdd(&ssum[(size_t)d * NH + h], v);
}

__global__ void edge_agg(const float* __restrict__ V, const float* __restrict__ ex,
                         const float* __restrict__ ssum, const int* __restrict__ src,
                         const int* __restrict__ dst, float* __restrict__ T,
                         float scale, int E64) {
  int t = blockIdx.x * blockDim.x + threadIdx.x;
  if (t >= E64) return;
  int e = t >> 6, c4 = t & 63;        // 4 floats per thread
  int h = c4 >> 3;
  int s = src[e], d = dst[e];
  float a = ex[(size_t)e * NH + h] / ssum[(size_t)d * NH + h] * scale;
  float4 v = *(const float4*)(V + (size_t)s * Dm + c4 * 4);
  float* tp = T + (size_t)d * Dm + c4 * 4;
  atomicAdd(tp + 0, v.x * a);
  atomicAdd(tp + 1, v.y * a);
  atomicAdd(tp + 2, v.z * a);
  atomicAdd(tp + 3, v.w * a);
}

// ---------------- skip + layernorm (one wave per row) ----------------

__global__ void skip_ln(const float* __restrict__ O, const float* __restrict__ Hn,
                        const float* __restrict__ skip, const float* __restrict__ g,
                        const float* __restrict__ b, float* __restrict__ out, int N) {
  int lane = threadIdx.x & 31;
  int wid  = threadIdx.x >> 5;
  int row  = blockIdx.x * 8 + wid;
  if (row >= N) return;
  float alpha = 1.f / (1.f + __expf(-skip[0]));
  float y[8], s = 0.f, s2 = 0.f;
#pragma unroll
  for (int j = 0; j < 8; j++) {
    int c = lane + 32 * j;
    float v = O[(size_t)row * Dm + c] * alpha + Hn[(size_t)row * Dm + c] * (1.f - alpha);
    y[j] = v; s += v; s2 += v * v;
  }
#pragma unroll
  for (int m = 16; m >= 1; m >>= 1) {
    s  += __shfl_xor(s, m, 32);
    s2 += __shfl_xor(s2, m, 32);
  }
  float mu  = s * (1.f / 256.f);
  float var = s2 * (1.f / 256.f) - mu * mu;
  float r   = rsqrtf(var + 1e-5f);
#pragma unroll
  for (int j = 0; j < 8; j++) {
    int c = lane + 32 * j;
    out[(size_t)row * Dm + c] = (y[j] - mu) * r * g[c] + b[c];
  }
}

// ---------------- host orchestration ----------------

extern "C" void kernel_launch(void* const* d_in, const int* in_sizes, int n_in,
                              void* d_out, int out_size, void* d_ws, size_t ws_size,
                              hipStream_t stream) {
  (void)n_in; (void)out_size; (void)ws_size;

  // setup_inputs() dict order
  const float* h_user = (const float*)d_in[0];
  const float* h_item = (const float*)d_in[1];
  const float* Wk_u = (const float*)d_in[2];  const float* bk_u = (const float*)d_in[3];
  const float* Wq_u = (const float*)d_in[4];  const float* bq_u = (const float*)d_in[5];
  const float* Wv_u = (const float*)d_in[6];  const float* bv_u = (const float*)d_in[7];
  const float* Wa_u = (const float*)d_in[8];  const float* ba_u = (const float*)d_in[9];
  const float* lng_u = (const float*)d_in[10]; const float* lnb_u = (const float*)d_in[11];
  const float* skip_u = (const float*)d_in[12];
  const float* Wk_i = (const float*)d_in[13]; const float* bk_i = (const float*)d_in[14];
  const float* Wq_i = (const float*)d_in[15]; const float* bq_i = (const float*)d_in[16];
  const float* Wv_i = (const float*)d_in[17]; const float* bv_i = (const float*)d_in[18];
  const float* Wa_i = (const float*)d_in[19]; const float* ba_i = (const float*)d_in[20];
  const float* lng_i = (const float*)d_in[21]; const float* lnb_i = (const float*)d_in[22];
  const float* skip_i = (const float*)d_in[23];
  const float* pri_uu = (const float*)d_in[24];
  const float* att_uu = (const float*)d_in[25];
  const float* msg_uu = (const float*)d_in[26];
  const int*   src_uu = (const int*)d_in[27];
  const int*   dst_uu = (const int*)d_in[28];
  const float* pri_ui = (const float*)d_in[29];
  const float* att_ui = (const float*)d_in[30];
  const float* msg_ui = (const float*)d_in[31];
  const int*   src_ui = (const int*)d_in[32];
  const int*   dst_ui = (const int*)d_in[33];
  const float* pri_iu = (const float*)d_in[34];
  const float* att_iu = (const float*)d_in[35];
  const float* msg_iu = (const float*)d_in[36];
  const int*   src_iu = (const int*)d_in[37];
  const int*   dst_iu = (const int*)d_in[38];

  const int NU = in_sizes[0] / Dm;
  const int NI = in_sizes[1] / Dm;
  const int E  = in_sizes[27];
  const int NMAX  = NU > NI ? NU : NI;
  const int NPAD  = (NMAX + 127) & ~127;   // async staging reads whole 128-row tiles

  // ---- workspace carve ----
  char* w = (char*)d_ws;
  auto carve = [&](size_t bytes) -> void* {
    void* p = (void*)w; w += (bytes + 255) & ~(size_t)255; return p;
  };
  const size_t WB = (size_t)Dm * Dm * sizeof(__bf16);
  __bf16* WqU_bf = (__bf16*)carve(WB);
  __bf16* WqI_bf = (__bf16*)carve(WB);
  __bf16* WaU_bf = (__bf16*)carve(WB);
  __bf16* WaI_bf = (__bf16*)carve(WB);
  __bf16* Wk_bf  = (__bf16*)carve(WB);   // per-relation, reused
  __bf16* Wv_bf  = (__bf16*)carve(WB);
  float* bk_eff = (float*)carve(Dm * sizeof(float));
  float* bv_eff = (float*)carve(Dm * sizeof(float));
  const size_t FEAT   = (size_t)NMAX * Dm * sizeof(float);
  const size_t FEATBF = (size_t)NPAD * Dm * sizeof(__bf16);
  __bf16* hU_bf = (__bf16*)carve(FEATBF);  // bf16 activations; reused for Tu later
  __bf16* hI_bf = (__bf16*)carve(FEATBF);  // reused for Ti later
  float* Qu   = (float*)carve(FEAT);
  float* Qi   = (float*)carve(FEAT);
  float* Kbuf = (float*)carve(FEAT);     // reused as O_user at the end
  float* Vbuf = (float*)carve(FEAT);     // reused as O_item at the end
  float* Tu   = (float*)carve(FEAT);
  float* Ti   = (float*)carve(FEAT);
  float* score = (float*)carve((size_t)E * NH * sizeof(float));
  float* ex    = (float*)carve((size_t)E * NH * sizeof(float));
  float* smax  = (float*)carve((size_t)NMAX * NH * sizeof(float));
  float* ssum  = (float*)carve((size_t)NMAX * NH * sizeof(float));

  const dim3 blk(256);
  auto gemm = [&](const __bf16* A, const __bf16* Wt, const float* bias, float* C, int M) {
    dim3 grid(Dm / 64, (M + 127) / 128);
    gemm_bf16<<<grid, blk, 0, stream>>>(A, Wt, bias, C, M);
  };

  // 1) weight conversions (transposed bf16)
  cvt_w_t<<<256, blk, 0, stream>>>(Wq_u, WqU_bf);
  cvt_w_t<<<256, blk, 0, stream>>>(Wq_i, WqI_bf);
  cvt_w_t<<<256, blk, 0, stream>>>(Wa_u, WaU_bf);
  cvt_w_t<<<256, blk, 0, stream>>>(Wa_i, WaI_bf);

  // 2) activations -> bf16 (natural K order)
  cvt_perm_bf16<<<4096, blk, 0, stream>>>(h_user, hU_bf, (long)NU * Dm);
  cvt_perm_bf16<<<4096, blk, 0, stream>>>(h_item, hI_bf, (long)NI * Dm);

  // 3) Q projections (persist across relations)
  gemm(hU_bf, WqU_bf, bq_u, Qu, NU);
  gemm(hI_bf, WqI_bf, bq_i, Qi, NI);

  // 4) zero aggregation targets
  zero_f<<<2048, blk, 0, stream>>>(Tu, (long)NU * Dm);
  zero_f<<<2048, blk, 0, stream>>>(Ti, (long)NI * Dm);

  // 5) relations
  struct Rel {
    const float *Wk, *bk, *Wv, *bv, *att, *msg, *pri;
    const int *src, *dst;
    const __bf16 *hsrc;
    const float *Q;
    int Nsrc, Ndst;
    float* T;
    float scale;
  };
  Rel rels[3] = {
    // (user -> user), target user, mean over 2 relations -> 0.5
    { Wk_u, bk_u, Wv_u, bv_u, att_uu, msg_uu, pri_uu, src_uu, dst_uu, hU_bf, Qu, NU, NU, Tu, 0.5f },
    // (item -> user), target user
    { Wk_i, bk_i, Wv_i, bv_i, att_iu, msg_iu, pri_iu, src_iu, dst_iu, hI_bf, Qu, NI, NU, Tu, 0.5f },
    // (user -> item), target item, single relation -> 1.0
    { Wk_u, bk_u, Wv_u, bv_u, att_ui, msg_ui, pri_ui, src_ui, dst_ui, hU_bf, Qi, NU, NI, Ti, 1.0f },
  };

  const int EH  = E * NH;
  const int E64 = E * 64;
  for (int r = 0; r < 3; r++) {
    const Rel& R = rels[r];
    // fold per-head relation transform into the projection weights
    compose_w_t<<<256, blk, 0, stream>>>(R.Wk, R.att, Wk_bf);
    compose_w_t<<<256, blk, 0, stream>>>(R.Wv, R.msg, Wv_bf);
    compose_b<<<1, blk, 0, stream>>>(R.bk, R.att, bk_eff);
    compose_b<<<1, blk, 0, stream>>>(R.bv, R.msg, bv_eff);
    // K/V projections via WMMA
    gemm(R.hsrc, Wk_bf, bk_eff, Kbuf, R.Nsrc);
    gemm(R.hsrc, Wv_bf, bv_eff, Vbuf, R.Nsrc);
    // segment softmax over dst + weighted scatter
    int nsm = R.Ndst * NH;
    init_softmax<<<(nsm + 255) / 256, blk, 0, stream>>>(smax, ssum, nsm);
    edge_score<<<(EH + 255) / 256, blk, 0, stream>>>(R.Q, Kbuf, R.src, R.dst, R.pri, score, smax, EH);
    edge_exp<<<(EH + 255) / 256, blk, 0, stream>>>(score, smax, R.dst, ex, ssum, EH);
    edge_agg<<<(E64 + 255) / 256, blk, 0, stream>>>(Vbuf, ex, ssum, R.src, R.dst, R.T, R.scale, E64);
  }

  // 6) aggregated features -> bf16 (reuse activation buffers), output projections
  cvt_perm_bf16<<<4096, blk, 0, stream>>>(Tu, hU_bf, (long)NU * Dm);
  cvt_perm_bf16<<<4096, blk, 0, stream>>>(Ti, hI_bf, (long)NI * Dm);
  gemm(hU_bf, WaU_bf, ba_u, Kbuf, NU);
  gemm(hI_bf, WaI_bf, ba_i, Vbuf, NI);

  // 7) skip + layernorm -> d_out (new_u then new_i, flat)
  float* out_u = (float*)d_out;
  float* out_i = out_u + (size_t)NU * Dm;
  skip_ln<<<(NU + 7) / 8, blk, 0, stream>>>(Kbuf, h_user, skip_u, lng_u, lnb_u, out_u, NU);
  skip_ln<<<(NI + 7) / 8, blk, 0, stream>>>(Vbuf, h_item, skip_i, lng_i, lnb_i, out_i, NI);
}
